// RefDissassembler_44530220925378
// MI455X (gfx1250) — matrix operand order repaired and checked by reading, not compiled, not run on hardware
//
#include <hip/hip_runtime.h>

// ---------------- problem constants ----------------
#define BB      8
#define SS      4096
#define HH      1024
#define SEGEX   64
#define NSEG    512          // BB * SEGEX
#define PP      4096         // pairs
#define KK      2048         // 2*HH (GEMM K)

typedef __attribute__((ext_vector_type(16))) __bf16 v16bf;
typedef __attribute__((ext_vector_type(8)))  float  v8f;

// ---------------------------------------------------------------------------
// CDNA5 async global->LDS copy: 32 bytes per lane (2 x b128, shared addr pair;
// the 24-bit inst offset applies to BOTH the LDS dest and the global source).
// Tracked by ASYNCcnt.
// ---------------------------------------------------------------------------
__device__ __forceinline__ unsigned lds_addr32(const void* p) {
    // LDS aperture occupies addr[63:32]; low 32 bits are the LDS byte offset.
    return (unsigned)(unsigned long long)p;
}
__device__ __forceinline__ void async_copy32B(unsigned lds, const void* g) {
    asm volatile("global_load_async_to_lds_b128 %0, %1, off\n\t"
                 "global_load_async_to_lds_b128 %0, %1, off offset:16"
                 :: "v"(lds), "v"(g) : "memory");
}
__device__ __forceinline__ void wait_async0() {
    asm volatile("s_wait_asynccnt 0x0" ::: "memory");
}

// ---------------------------------------------------------------------------
// K0: W1 [2048][1024] f32  ->  W1t [1024][2048] bf16  (transpose + convert)
// ---------------------------------------------------------------------------
__global__ void w1_convert_kernel(const float* __restrict__ W1,
                                  __bf16* __restrict__ W1t) {
    int idx = blockIdx.x * blockDim.x + threadIdx.x;   // 0 .. 2048*1024-1
    int k = idx >> 10;          // row in W1   (K dim)
    int n = idx & (HH - 1);     // col in W1   (N dim)
    W1t[(size_t)n * KK + k] = (__bf16)W1[idx];
}

// ---------------------------------------------------------------------------
// K1: span mean pooling.  One block per global segment (512 blocks).
// segment_ids are sorted within each example row -> contiguous token range.
// ---------------------------------------------------------------------------
__global__ void span_pool_kernel(const float* __restrict__ seq,
                                 const int*   __restrict__ seg_ids,
                                 __bf16* __restrict__ span_bf) {
    const int gseg = blockIdx.x;       // 0..511
    const int ex   = gseg >> 6;        // example
    const int ls   = gseg & (SEGEX - 1);

    __shared__ int s_start, s_end;
    if (threadIdx.x == 0) {
        const int* row = seg_ids + ex * SS;
        int lo = 0, hi = SS;
        while (lo < hi) { int mid = (lo + hi) >> 1; if (row[mid] <  ls) lo = mid + 1; else hi = mid; }
        s_start = lo;
        hi = SS;
        while (lo < hi) { int mid = (lo + hi) >> 1; if (row[mid] <= ls) lo = mid + 1; else hi = mid; }
        s_end = lo;
    }
    __syncthreads();

    const int start = s_start, end = s_end;
    const int cnt   = end - start;
    const float inv = (cnt > 0) ? (1.0f / (float)cnt) : 0.0f;   // empty seg -> 0

    const float* base = seq + (size_t)ex * SS * HH;
    const int ch = threadIdx.x * 4;                     // 256 thr * 4 = 1024
    float4 acc = make_float4(0.f, 0.f, 0.f, 0.f);
    for (int t = start; t < end; ++t) {
        const float4 v = *(const float4*)(base + (size_t)t * HH + ch);
        acc.x += v.x; acc.y += v.y; acc.z += v.z; acc.w += v.w;
    }
    __bf16* out = span_bf + (size_t)gseg * HH + ch;
    out[0] = (__bf16)(acc.x * inv);
    out[1] = (__bf16)(acc.y * inv);
    out[2] = (__bf16)(acc.z * inv);
    out[3] = (__bf16)(acc.w * inv);
}

// ---------------------------------------------------------------------------
// K2: fused pair-gather + GEMM(relu(pairs@W1+b1)) + dot(W2) partial epilogue.
// Grid: (32 M-blocks, 8 N-blocks), 256 threads = 8 waves (wave32).
// Block tile: M=128, N=128, K-step 32.  Wave tile: M=32, N=64 (2x4 WMMA).
// Double-buffered LDS; next tile staged via async global->LDS b128 copies
// issued before the current slice's WMMAs (latency hidden under the math).
// ---------------------------------------------------------------------------
__global__ void relation_gemm_kernel(const __bf16* __restrict__ span_bf,
                                     const __bf16* __restrict__ W1t,
                                     const int*    __restrict__ pair_idx,
                                     const float*  __restrict__ b1,
                                     const float*  __restrict__ W2,
                                     float* __restrict__ partial) {
    const int mb = blockIdx.x;          // 0..31
    const int nb = blockIdx.y;          // 0..7
    const int m0 = mb * 128;
    const int n0 = nb * 128;

    const int tid     = threadIdx.x;
    const int wave    = tid >> 5;       // 0..7
    const int lane    = tid & 31;
    const int lane_lo = lane & 15;      // 0..15
    const int lane_hi = lane >> 4;      // 0..1
    const int wm      = wave & 3;       // M wave index 0..3
    const int wn      = wave >> 2;      // N wave index 0..1

    __shared__ __bf16 sA[2][128][32];   // A tiles (pair rows), double buffered
    __shared__ __bf16 sB[2][128][32];   // B tiles, N-major [n][k]
    __shared__ float  sPart[128][2];    // per-row partial, per N-wave

    // staging: thread t moves 32B of A and 32B of B: row = t/2, half = t&1
    const int srow  = tid >> 1;
    const int shalf = tid & 1;          // element offset shalf*16 (32B chunk)

    // loop-invariant gather bases (pair rows never change)
    const int prow0 = pair_idx[(m0 + srow) * 2 + 0];
    const int prow1 = pair_idx[(m0 + srow) * 2 + 1];
    const __bf16* __restrict__ aptr0 = span_bf + (size_t)prow0 * HH + shalf * 16;
    const __bf16* __restrict__ aptr1 = span_bf + (size_t)prow1 * HH + shalf * 16;
    const __bf16* __restrict__ bptr  = W1t + (size_t)(n0 + srow) * KK + shalf * 16;

    const unsigned ldsA0 = lds_addr32(&sA[0][srow][shalf * 16]);
    const unsigned ldsB0 = lds_addr32(&sB[0][srow][shalf * 16]);
    const unsigned bufStride = (unsigned)(128 * 32 * sizeof(__bf16));   // 8KB

    v8f acc[2][4];
#pragma unroll
    for (int fm = 0; fm < 2; ++fm)
#pragma unroll
        for (int fn = 0; fn < 4; ++fn)
            acc[fm][fn] = (v8f){0.f, 0.f, 0.f, 0.f, 0.f, 0.f, 0.f, 0.f};

    // ---- prologue: stage K-slice 0 into buffer 0 (async) ----
    async_copy32B(ldsA0, aptr0);
    async_copy32B(ldsB0, bptr);
    wait_async0();
    __syncthreads();

    for (int ki = 0; ki < KK / 32; ++ki) {
        const int cur = ki & 1;

        // ---- issue next K-slice's async copies into the other buffer ----
        // (safe: the barrier ending iteration ki-1 guaranteed all waves
        //  finished their fragment reads of buffer cur^1)
        if (ki + 1 < KK / 32) {
            const int k0n = (ki + 1) * 32;
            const __bf16* ap = ((k0n >= HH) ? aptr1 : aptr0) + (k0n & (HH - 1));
            const unsigned nb_off = (unsigned)(cur ^ 1) * bufStride;
            async_copy32B(ldsA0 + nb_off, ap);
            async_copy32B(ldsB0 + nb_off, bptr + k0n);
        }

        // ---- build A fragments (16x32 bf16 ISA layout) ----
        v16bf afrag[2];
#pragma unroll
        for (int fm = 0; fm < 2; ++fm) {
            const __bf16* ap2 = &sA[cur][wm * 32 + fm * 16 + lane_lo][0];
#pragma unroll
            for (int i = 0; i < 8; ++i) {
                afrag[fm][i]     = ap2[lane_hi * 8 + i];          // K = hi*8 + i
                afrag[fm][8 + i] = ap2[16 + lane_hi * 8 + i];     // K = 16 + hi*8 + i
            }
        }
        // ---- build B fragments (32x16 bf16 ISA layout, N-major LDS) ----
        v16bf bfrag[4];
#pragma unroll
        for (int fn = 0; fn < 4; ++fn) {
            const __bf16* bp2 = &sB[cur][wn * 64 + fn * 16 + lane_lo][0];
#pragma unroll
            for (int i = 0; i < 16; ++i)
                bfrag[fn][i] = bp2[(i < 8) ? (lane_hi * 8 + i) : (16 + lane_hi * 8 + (i - 8))];  // K = hi*8 + i (i<8), 16 + hi*8 + (i-8) (i>=8)
        }

        // ---- 8 WMMAs per wave per K-slice ----
#pragma unroll
        for (int fm = 0; fm < 2; ++fm)
#pragma unroll
            for (int fn = 0; fn < 4; ++fn)
                acc[fm][fn] = __builtin_amdgcn_wmma_f32_16x16x32_bf16(
                    false, afrag[fm], false, bfrag[fn],
                    (short)0, acc[fm][fn], false, false);

        // ---- make next tile visible to all waves ----
        if (ki + 1 < KK / 32) wait_async0();
        __syncthreads();
    }

    // ---- epilogue: +b1, ReLU, dot with W2 over this block's 128 N columns ----
    float part[2][8];
#pragma unroll
    for (int fm = 0; fm < 2; ++fm)
#pragma unroll
        for (int r = 0; r < 8; ++r) part[fm][r] = 0.f;

#pragma unroll
    for (int fn = 0; fn < 4; ++fn) {
        const int gn   = n0 + wn * 64 + fn * 16 + lane_lo;  // C col = lane_lo
        const float bv = b1[gn];
        const float wv = W2[gn];
#pragma unroll
        for (int fm = 0; fm < 2; ++fm)
#pragma unroll
            for (int r = 0; r < 8; ++r) {
                float h = acc[fm][fn][r] + bv;
                h = h > 0.f ? h : 0.f;
                part[fm][r] += h * wv;                      // C row = hi*8 + r
            }
    }
    // reduce across the 16 lanes that share a row (fixed order -> deterministic)
#pragma unroll
    for (int m = 1; m < 16; m <<= 1)
#pragma unroll
        for (int fm = 0; fm < 2; ++fm)
#pragma unroll
            for (int r = 0; r < 8; ++r)
                part[fm][r] += __shfl_xor(part[fm][r], m, 32);

    if (lane_lo == 0) {
#pragma unroll
        for (int fm = 0; fm < 2; ++fm)
#pragma unroll
            for (int r = 0; r < 8; ++r) {
                const int row = wm * 32 + fm * 16 + lane_hi * 8 + r;
                sPart[row][wn] = part[fm][r];
            }
    }
    __syncthreads();
    if (tid < 128)
        partial[(size_t)nb * PP + m0 + tid] = sPart[tid][0] + sPart[tid][1];
}

// ---------------------------------------------------------------------------
// K3: deterministic reduction of the 8 N-block partials + b2
// ---------------------------------------------------------------------------
__global__ void logits_reduce_kernel(const float* __restrict__ partial,
                                     const float* __restrict__ b2,
                                     float* __restrict__ out) {
    const int p = blockIdx.x * blockDim.x + threadIdx.x;
    float s = b2[0];
#pragma unroll
    for (int nbk = 0; nbk < 8; ++nbk) s += partial[(size_t)nbk * PP + p];
    out[p] = s;
}

// ---------------------------------------------------------------------------
extern "C" void kernel_launch(void* const* d_in, const int* in_sizes, int n_in,
                              void* d_out, int out_size, void* d_ws, size_t ws_size,
                              hipStream_t stream) {
    const float* seq      = (const float*)d_in[0];  // [8,4096,1024]
    const int*   seg_ids  = (const int*)  d_in[1];  // [8,4096]
    const int*   pair_idx = (const int*)  d_in[2];  // [4096,2]
    const float* W1       = (const float*)d_in[3];  // [2048,1024]
    const float* b1       = (const float*)d_in[4];  // [1024]
    const float* W2       = (const float*)d_in[5];  // [1024,1]
    const float* b2       = (const float*)d_in[6];  // [1]
    float* out = (float*)d_out;                     // [4096,1]

    // workspace layout
    char* ws = (char*)d_ws;
    __bf16* span_bf = (__bf16*)ws;                                   // 1 MB
    __bf16* W1t     = (__bf16*)(ws + (size_t)NSEG * HH * 2);         // 4 MB
    float*  partial = (float*)(ws + (size_t)NSEG * HH * 2
                                  + (size_t)HH * KK * 2);            // 128 KB

    w1_convert_kernel<<<(KK * HH) / 256, 256, 0, stream>>>(W1, W1t);
    span_pool_kernel<<<NSEG, 256, 0, stream>>>(seq, seg_ids, span_bf);
    relation_gemm_kernel<<<dim3(PP / 128, HH / 128), 256, 0, stream>>>(
        span_bf, W1t, pair_idx, b1, W2, partial);
    logits_reduce_kernel<<<PP / 256, 256, 0, stream>>>(partial, b2, out);
}
